// MaskedAutoencoderViT_85366769975326
// MI455X (gfx1250) — compile-verified, matrix operand order checked by reading, not run on hardware
//
#include <hip/hip_runtime.h>
#include <hip/hip_bf16.h>
#include <math.h>
#include <stdint.h>

// ---------------- problem constants ----------------
#define BATCH   16
#define GH      12
#define GW      25
#define LPATCH  300      // GH*GW
#define PATCHK  100      // conv kernel width (P=1, Q=100)
#define DMODEL  768
#define DHALF   384
#define NHEADS  12
#define HDIM    64
#define NTOK    76       // 1 + LEN_KEEP
#define LKEEP   75
#define TPB     80       // tokens padded to multiple of 16 for WMMA
#define MLPDIM  3072
#define SCALE_F 0.125f   // 64^-0.5

// ---------------- WMMA types ----------------
typedef __attribute__((ext_vector_type(16))) __bf16 v16bf;
typedef __attribute__((ext_vector_type(8)))  __bf16 v8bf;
typedef __attribute__((ext_vector_type(4)))  __bf16 v4bf;
typedef __attribute__((ext_vector_type(8)))  float  v8f;

__device__ __forceinline__ __bf16 f2bf(float f) {
  // round-to-nearest-even fp32 -> bf16
  unsigned u = __float_as_uint(f);
  u += 0x7FFFu + ((u >> 16) & 1u);
  unsigned short h = (unsigned short)(u >> 16);
  return __builtin_bit_cast(__bf16, h);
}

// elementwise fp32 -> bf16 (4 elements / thread)
__global__ void cvt_f32_bf16(const float* __restrict__ in, __bf16* __restrict__ out, int n4) {
  int i = blockIdx.x * 256 + threadIdx.x;
  if (i < n4) {
    float4 v = ((const float4*)in)[i];
    v4bf o;
    o[0] = f2bf(v.x); o[1] = f2bf(v.y); o[2] = f2bf(v.z); o[3] = f2bf(v.w);
    ((v4bf*)out)[i] = o;
  }
}

// ======================================================================
// GEMM: out[M,N] = A[M,K](bf16) @ W[N,K](bf16)^T (+bias)(+res)(+GELU)
// One wave per 16x64 macro-tile: 1 A-fragment feeds 4 WMMAs.
// grid = (M/16, N/64), block = 32 (wave32, EXEC all-ones around WMMA).
// Inner loop: pure b128 loads + v_wmma, no conversion VALU.
// ======================================================================
__global__ __launch_bounds__(32)
void wmma_gemm_bf16(const __bf16* __restrict__ A, int lda,
                    const __bf16* __restrict__ W, int ldw,
                    const float* __restrict__ bias,
                    const float* __restrict__ res,
                    void* __restrict__ outp, int ldo,
                    int K, int fuse_gelu, int out_bf16)
{
  const int lane = threadIdx.x;
  const int mrow = lane & 15;   // A-row / W-row / C-column owned by this lane
  const int half = lane >> 4;   // K-half selector (A/B), M-offset (C)
  const int m0 = blockIdx.x * 16;
  const int n0 = blockIdx.y * 64;

  const __bf16* arow  = A + (size_t)(m0 + mrow) * lda;
  const __bf16* wrow0 = W + (size_t)(n0 + mrow) * ldw;
  const __bf16* wrow1 = wrow0 + (size_t)16 * ldw;
  const __bf16* wrow2 = wrow0 + (size_t)32 * ldw;
  const __bf16* wrow3 = wrow0 + (size_t)48 * ldw;

  v8f c0 = {}, c1 = {}, c2 = {}, c3 = {};
  if (res) {
#pragma unroll
    for (int j = 0; j < 8; ++j) {
      size_t r = (size_t)(m0 + j + 8 * half) * ldo + n0 + mrow;
      c0[j] = res[r];      c1[j] = res[r + 16];
      c2[j] = res[r + 32]; c3[j] = res[r + 48];
    }
  }

  for (int kb = 0; kb < K; kb += 32) {
    // ISA 16-bit operand layout: lane holds K = {8h+0..7} and {16+8h+0..7}
    const int klo = kb + 8 * half;
    v8bf al  = *(const v8bf*)(arow  + klo);
    v8bf ah  = *(const v8bf*)(arow  + klo + 16);
    v8bf w0l = *(const v8bf*)(wrow0 + klo);
    v8bf w0h = *(const v8bf*)(wrow0 + klo + 16);
    v8bf w1l = *(const v8bf*)(wrow1 + klo);
    v8bf w1h = *(const v8bf*)(wrow1 + klo + 16);
    v8bf w2l = *(const v8bf*)(wrow2 + klo);
    v8bf w2h = *(const v8bf*)(wrow2 + klo + 16);
    v8bf w3l = *(const v8bf*)(wrow3 + klo);
    v8bf w3h = *(const v8bf*)(wrow3 + klo + 16);

    if (kb + 32 < K) {
      __builtin_prefetch(arow  + kb + 32, 0, 3);   // global_prefetch_b8, near caches
      __builtin_prefetch(wrow0 + kb + 32, 0, 3);
      __builtin_prefetch(wrow1 + kb + 32, 0, 3);
      __builtin_prefetch(wrow2 + kb + 32, 0, 3);
      __builtin_prefetch(wrow3 + kb + 32, 0, 3);
    }

    v16bf av = __builtin_shufflevector(al, ah, 0,1,2,3,4,5,6,7,8,9,10,11,12,13,14,15);
    v16bf wv;
    wv = __builtin_shufflevector(w0l, w0h, 0,1,2,3,4,5,6,7,8,9,10,11,12,13,14,15);
    c0 = __builtin_amdgcn_wmma_f32_16x16x32_bf16(false, av, false, wv, (short)0, c0, false, false);
    wv = __builtin_shufflevector(w1l, w1h, 0,1,2,3,4,5,6,7,8,9,10,11,12,13,14,15);
    c1 = __builtin_amdgcn_wmma_f32_16x16x32_bf16(false, av, false, wv, (short)0, c1, false, false);
    wv = __builtin_shufflevector(w2l, w2h, 0,1,2,3,4,5,6,7,8,9,10,11,12,13,14,15);
    c2 = __builtin_amdgcn_wmma_f32_16x16x32_bf16(false, av, false, wv, (short)0, c2, false, false);
    wv = __builtin_shufflevector(w3l, w3h, 0,1,2,3,4,5,6,7,8,9,10,11,12,13,14,15);
    c3 = __builtin_amdgcn_wmma_f32_16x16x32_bf16(false, av, false, wv, (short)0, c3, false, false);
  }

  float b0 = 0.f, b1 = 0.f, b2 = 0.f, b3 = 0.f;
  if (bias) {
    b0 = bias[n0 + mrow];      b1 = bias[n0 + 16 + mrow];
    b2 = bias[n0 + 32 + mrow]; b3 = bias[n0 + 48 + mrow];
  }
#pragma unroll
  for (int j = 0; j < 8; ++j) {
    size_t r = (size_t)(m0 + j + 8 * half) * ldo + n0 + mrow;
    float v0 = c0[j] + b0, v1 = c1[j] + b1, v2 = c2[j] + b2, v3 = c3[j] + b3;
    if (fuse_gelu) {
      v0 = 0.5f * v0 * (1.0f + erff(v0 * 0.70710678118654752f));
      v1 = 0.5f * v1 * (1.0f + erff(v1 * 0.70710678118654752f));
      v2 = 0.5f * v2 * (1.0f + erff(v2 * 0.70710678118654752f));
      v3 = 0.5f * v3 * (1.0f + erff(v3 * 0.70710678118654752f));
    }
    if (out_bf16) {
      __bf16* o = (__bf16*)outp;
      o[r] = f2bf(v0); o[r + 16] = f2bf(v1); o[r + 32] = f2bf(v2); o[r + 48] = f2bf(v3);
    } else {
      float* o = (float*)outp;
      o[r] = v0; o[r + 16] = v1; o[r + 32] = v2; o[r + 48] = v3;
    }
  }
}

// ======================================================================
// Patch embed + positional embeddings. One block per (b, l).
// The 100-float patch is staged into LDS with GLOBAL_LOAD_ASYNC_TO_LDS_B128
// (CDNA5 async data path, tracked by ASYNCcnt), drained with s_wait_asynccnt.
// ======================================================================
__global__ void patch_embed(const float* __restrict__ x, const float* __restrict__ am,
                            const int* __restrict__ yids,
                            const float* __restrict__ cw, const float* __restrict__ cb,
                            const float* __restrict__ px, const float* __restrict__ pyt,
                            float* __restrict__ hf)
{
  int l = blockIdx.x % LPATCH;
  int b = blockIdx.x / LPATCH;
  int gh = l / GW, gw = l % GW;
  __shared__ __align__(16) float patch[PATCHK + 4];
  int tid = threadIdx.x;  // 128
  if (tid < PATCHK / 4) {
    // per-lane: LDS[vdst] = MEM[vaddr], 16 bytes (GV mode, saddr = off)
    unsigned lds_addr = (unsigned)(uintptr_t)(&patch[tid * 4]);
    unsigned long long gaddr =
        (unsigned long long)(uintptr_t)(&x[((size_t)b * GH + gh) * 2500 + gw * PATCHK + tid * 4]);
    asm volatile("global_load_async_to_lds_b128 %0, %1, off"
                 :: "v"(lds_addr), "v"(gaddr) : "memory");
  }
  asm volatile("s_wait_asynccnt 0x0" ::: "memory");
  __syncthreads();

  float mask = am[(size_t)b * LPATCH + l];
  int yid = yids[(size_t)b * LPATCH + l];
  for (int d = tid; d < DMODEL; d += 128) {
    float acc = cb[d];
    const float* wr = cw + (size_t)d * PATCHK;
    for (int q = 0; q < PATCHK; ++q) acc += patch[q] * wr[q];
    if (d >= DHALF) acc += mask * px[(size_t)(gw + 1) * DHALF + (d - DHALF)];
    else            acc += pyt[(size_t)yid * DHALF + d];
    hf[((size_t)b * LPATCH + l) * DMODEL + d] = acc;
  }
}

// ======================================================================
// Stable argsort ranks + keep ids (rank order) + attention mask vector m
// (mask built from *sorted* kept indices, per the reference).
// ======================================================================
__global__ void shuffle_kernel(const float* __restrict__ noise, const float* __restrict__ am,
                               int* __restrict__ ids_keep, float* __restrict__ mvec)
{
  int b = blockIdx.x;
  int tid = threadIdx.x;  // 256
  __shared__ float ns[LPATCH];
  __shared__ int rk[LPATCH];
  for (int i = tid; i < LPATCH; i += 256) ns[i] = noise[(size_t)b * LPATCH + i];
  __syncthreads();
  for (int i = tid; i < LPATCH; i += 256) {
    float ni = ns[i];
    int r = 0;
    for (int j = 0; j < LPATCH; ++j) {
      float nj = ns[j];
      r += (nj < ni) || (nj == ni && j < i);
    }
    rk[i] = r;
    if (r < LKEEP) ids_keep[b * LKEEP + r] = i;   // rank order (for gather)
  }
  __syncthreads();
  for (int i = tid; i < LPATCH; i += 256) {
    if (rk[i] < LKEEP) {
      int pos = 0;
      for (int j = 0; j < i; ++j) pos += (rk[j] < LKEEP);  // sorted position
      mvec[b * TPB + 1 + pos] = am[(size_t)b * LPATCH + i];
    }
  }
  if (tid == 0) mvec[b * TPB] = 1.0f;
  if (tid >= NTOK && tid < TPB) mvec[b * TPB + tid] = 0.0f;
}

// ======================================================================
// Assemble padded token matrix h[B, TPB, D] (cls + gathered patches + pad)
// ======================================================================
__global__ void assemble_tokens(const float* __restrict__ hf, const int* __restrict__ ids_keep,
                                const float* __restrict__ cls, const float* __restrict__ px,
                                float* __restrict__ h)
{
  int t = blockIdx.x % TPB, b = blockIdx.x / TPB;
  int tid = threadIdx.x;  // 256
  float* dst = h + (size_t)(b * TPB + t) * DMODEL;
  if (t == 0) {
    for (int d = tid; d < DMODEL; d += 256)
      dst[d] = cls[d] + (d >= DHALF ? px[d - DHALF] : 0.0f);  // pex[0,0,:]
  } else if (t <= LKEEP) {
    int src = ids_keep[b * LKEEP + (t - 1)];
    const float* s = hf + ((size_t)b * LPATCH + src) * DMODEL;
    for (int d = tid; d < DMODEL; d += 256) dst[d] = s[d];
  } else {
    for (int d = tid; d < DMODEL; d += 256) dst[d] = 0.0f;
  }
}

// ======================================================================
// LayerNorm over last dim; optional bf16 output (feeds WMMA GEMMs).
// ======================================================================
__global__ void ln_kernel(const float* __restrict__ in, const float* __restrict__ w,
                          const float* __restrict__ b, void* __restrict__ outp,
                          int ntok, int tpb_in, int tpb_out, int out_bf16)
{
  int row = blockIdx.x;
  int bb = row / ntok, t = row % ntok;
  const float* x = in + (size_t)(bb * tpb_in + t) * DMODEL;
  size_t obase = (size_t)(bb * tpb_out + t) * DMODEL;
  int tid = threadIdx.x;  // 256
  __shared__ float red[256];
  float s = 0.0f;
  for (int d = tid; d < DMODEL; d += 256) s += x[d];
  red[tid] = s; __syncthreads();
  for (int wd = 128; wd > 0; wd >>= 1) { if (tid < wd) red[tid] += red[tid + wd]; __syncthreads(); }
  float mu = red[0] / DMODEL;
  __syncthreads();
  float v = 0.0f;
  for (int d = tid; d < DMODEL; d += 256) { float t2 = x[d] - mu; v += t2 * t2; }
  red[tid] = v; __syncthreads();
  for (int wd = 128; wd > 0; wd >>= 1) { if (tid < wd) red[tid] += red[tid + wd]; __syncthreads(); }
  float rstd = rsqrtf(red[0] / DMODEL + 1e-5f);
  if (out_bf16) {
    __bf16* y = (__bf16*)outp;
    for (int d = tid; d < DMODEL; d += 256)
      y[obase + d] = f2bf((x[d] - mu) * rstd * w[d] + b[d]);
  } else {
    float* y = (float*)outp;
    for (int d = tid; d < DMODEL; d += 256)
      y[obase + d] = (x[d] - mu) * rstd * w[d] + b[d];
  }
}

// ---- order-preserving float<->uint for atomic global max ----
__device__ __forceinline__ unsigned ord_enc(float f) {
  unsigned u = __float_as_uint(f);
  return (u & 0x80000000u) ? ~u : (u | 0x80000000u);
}
__device__ __forceinline__ float ord_dec(unsigned e) {
  unsigned u = (e & 0x80000000u) ? (e ^ 0x80000000u) : ~e;
  return __uint_as_float(u);
}

__global__ void set_u32(unsigned* p, unsigned v) { *p = v; }

// ======================================================================
// att[b,h,i,j] = scale * q.k (masked to -inf) ; global max reduction.
// ======================================================================
__global__ void attn_scores(const float* __restrict__ qkv, const float* __restrict__ mvec,
                            float* __restrict__ att, unsigned* __restrict__ gmax)
{
  int i = blockIdx.x, h = blockIdx.y, b = blockIdx.z;
  int tid = threadIdx.x;  // 128
  __shared__ float qs[HDIM];
  __shared__ float red[128];
  const float* qrow = qkv + (size_t)(b * TPB + i) * (3 * DMODEL) + h * HDIM;
  if (tid < HDIM) qs[tid] = qrow[tid];
  __syncthreads();
  float s = -INFINITY;
  if (tid < NTOK) {
    const float* krow = qkv + (size_t)(b * TPB + tid) * (3 * DMODEL) + DMODEL + h * HDIM;
    float acc = 0.0f;
    for (int d = 0; d < HDIM; ++d) acc += qs[d] * krow[d];
    acc *= SCALE_F;
    if (mvec[b * TPB + i] * mvec[b * TPB + tid] == 0.0f) acc = -INFINITY;
    att[(((size_t)(b * NHEADS) + h) * NTOK + i) * NTOK + tid] = acc;
    s = acc;
  }
  red[tid] = s;
  __syncthreads();
  for (int wd = 64; wd > 0; wd >>= 1) { if (tid < wd) red[tid] = fmaxf(red[tid], red[tid + wd]); __syncthreads(); }
  if (tid == 0) atomicMax(gmax, ord_enc(red[0]));
}

// exp(att - global_max), row-normalize with +1e-9 (reference semantics)
__global__ void attn_softmax(float* __restrict__ att, const unsigned* __restrict__ gmax)
{
  int i = blockIdx.x, h = blockIdx.y, b = blockIdx.z;
  int tid = threadIdx.x;  // 128
  __shared__ float red[128];
  float gm = ord_dec(*gmax);
  size_t base = (((size_t)(b * NHEADS) + h) * NTOK + i) * NTOK;
  float e = 0.0f;
  if (tid < NTOK) { float a = att[base + tid]; e = expf(a - gm); }
  red[tid] = e;
  __syncthreads();
  for (int wd = 64; wd > 0; wd >>= 1) { if (tid < wd) red[tid] += red[tid + wd]; __syncthreads(); }
  float inv = 1.0f / (red[0] + 1e-9f);
  if (tid < NTOK) att[base + tid] = e * inv;
}

// o[b,t,:] = att @ v, written as bf16 (feeds proj WMMA GEMM)
__global__ void attn_av(const float* __restrict__ att, const float* __restrict__ qkv,
                        __bf16* __restrict__ o)
{
  int t = blockIdx.x, b = blockIdx.y;
  int tid = threadIdx.x;  // 256
  for (int d = tid; d < DMODEL; d += 256) {
    float acc = 0.0f;
    if (t < NTOK) {
      int h = d / HDIM, dd = d % HDIM;
      size_t abase = (((size_t)(b * NHEADS) + h) * NTOK + t) * NTOK;
      for (int j = 0; j < NTOK; ++j)
        acc += att[abase + j] *
               qkv[(size_t)(b * TPB + j) * (3 * DMODEL) + 2 * DMODEL + h * HDIM + dd];
    }
    o[(size_t)(b * TPB + t) * DMODEL + d] = f2bf(acc);
  }
}

// ======================================================================
extern "C" void kernel_launch(void* const* d_in, const int* in_sizes, int n_in,
                              void* d_out, int out_size, void* d_ws, size_t ws_size,
                              hipStream_t stream) {
  (void)in_sizes; (void)n_in; (void)out_size; (void)ws_size;
  const float* x         = (const float*)d_in[0];
  const float* attn_mask = (const float*)d_in[1];
  const int*   yids      = (const int*)  d_in[2];
  const float* noise     = (const float*)d_in[3];
  const float* conv_w    = (const float*)d_in[4];
  const float* conv_b    = (const float*)d_in[5];
  const float* pos_x     = (const float*)d_in[6];
  const float* pos_y     = (const float*)d_in[7];
  const float* cls       = (const float*)d_in[8];
  const float* qkv_w     = (const float*)d_in[9];
  const float* qkv_b     = (const float*)d_in[10];
  const float* proj_w    = (const float*)d_in[11];
  const float* proj_b    = (const float*)d_in[12];
  const float* ln1_w     = (const float*)d_in[13];
  const float* ln1_b     = (const float*)d_in[14];
  const float* ln2_w     = (const float*)d_in[15];
  const float* ln2_b     = (const float*)d_in[16];
  const float* fc1_w     = (const float*)d_in[17];
  const float* fc1_b     = (const float*)d_in[18];
  const float* fc2_w     = (const float*)d_in[19];
  const float* fc2_b     = (const float*)d_in[20];
  const float* norm_w    = (const float*)d_in[21];
  const float* norm_b    = (const float*)d_in[22];

  char* wsp = (char*)d_ws;
  size_t off = 0;
  auto carve = [&](size_t nbytes) -> char* {
    char* p = wsp + off;
    off += (nbytes + 255) & ~(size_t)255;
    return p;
  };
  float*    hf       = (float*)carve((size_t)BATCH * LPATCH * DMODEL * 4);
  int*      ids_keep = (int*)  carve((size_t)BATCH * LKEEP * 4);
  float*    mvec     = (float*)carve((size_t)BATCH * TPB * 4);
  float*    hbuf     = (float*)carve((size_t)BATCH * TPB * DMODEL * 4);
  __bf16*   ybuf     = (__bf16*)carve((size_t)BATCH * TPB * DMODEL * 2);
  float*    qkvb     = (float*)carve((size_t)BATCH * TPB * 3 * DMODEL * 4);
  float*    attb     = (float*)carve((size_t)BATCH * NHEADS * NTOK * NTOK * 4);
  __bf16*   obuf     = (__bf16*)carve((size_t)BATCH * TPB * DMODEL * 2);
  __bf16*   mlpb     = (__bf16*)carve((size_t)BATCH * TPB * MLPDIM * 2);
  // per-layer bf16 weight staging (reused every layer)
  __bf16*   qw_bf    = (__bf16*)carve((size_t)3 * DMODEL * DMODEL * 2);
  __bf16*   pw_bf    = (__bf16*)carve((size_t)DMODEL * DMODEL * 2);
  __bf16*   f1w_bf   = (__bf16*)carve((size_t)MLPDIM * DMODEL * 2);
  __bf16*   f2w_bf   = (__bf16*)carve((size_t)DMODEL * MLPDIM * 2);
  unsigned* gmax     = (unsigned*)carve(4);

  const int Mrows = BATCH * TPB;  // 1280, multiple of 16

  patch_embed<<<BATCH * LPATCH, 128, 0, stream>>>(x, attn_mask, yids, conv_w, conv_b,
                                                  pos_x, pos_y, hf);
  shuffle_kernel<<<BATCH, 256, 0, stream>>>(noise, attn_mask, ids_keep, mvec);
  assemble_tokens<<<BATCH * TPB, 256, 0, stream>>>(hf, ids_keep, cls, pos_x, hbuf);

  const int n4_qkv  = (3 * DMODEL * DMODEL) / 4;   // 442368
  const int n4_proj = (DMODEL * DMODEL) / 4;       // 147456
  const int n4_fc   = (MLPDIM * DMODEL) / 4;       // 589824

  for (int l = 0; l < 12; ++l) {
    const float* qw  = qkv_w  + (size_t)l * 3 * DMODEL * DMODEL;
    const float* qb  = qkv_b  + (size_t)l * 3 * DMODEL;
    const float* pw  = proj_w + (size_t)l * DMODEL * DMODEL;
    const float* pb  = proj_b + (size_t)l * DMODEL;
    const float* f1w = fc1_w  + (size_t)l * MLPDIM * DMODEL;
    const float* f1b = fc1_b  + (size_t)l * MLPDIM;
    const float* f2w = fc2_w  + (size_t)l * DMODEL * MLPDIM;
    const float* f2b = fc2_b  + (size_t)l * DMODEL;

    // stage this layer's weights as bf16
    cvt_f32_bf16<<<n4_qkv / 256, 256, 0, stream>>>(qw, qw_bf, n4_qkv);
    cvt_f32_bf16<<<n4_proj / 256, 256, 0, stream>>>(pw, pw_bf, n4_proj);
    cvt_f32_bf16<<<n4_fc / 256, 256, 0, stream>>>(f1w, f1w_bf, n4_fc);
    cvt_f32_bf16<<<n4_fc / 256, 256, 0, stream>>>(f2w, f2w_bf, n4_fc);

    ln_kernel<<<Mrows, 256, 0, stream>>>(hbuf, ln1_w + l * DMODEL, ln1_b + l * DMODEL,
                                         ybuf, TPB, TPB, TPB, 1);
    wmma_gemm_bf16<<<dim3(Mrows / 16, (3 * DMODEL) / 64), 32, 0, stream>>>(
        ybuf, DMODEL, qw_bf, DMODEL, qb, nullptr, qkvb, 3 * DMODEL, DMODEL, 0, 0);

    set_u32<<<1, 1, 0, stream>>>(gmax, 0u);
    attn_scores<<<dim3(NTOK, NHEADS, BATCH), 128, 0, stream>>>(qkvb, mvec, attb, gmax);
    attn_softmax<<<dim3(NTOK, NHEADS, BATCH), 128, 0, stream>>>(attb, gmax);
    attn_av<<<dim3(TPB, BATCH), 256, 0, stream>>>(attb, qkvb, obuf);

    wmma_gemm_bf16<<<dim3(Mrows / 16, DMODEL / 64), 32, 0, stream>>>(
        obuf, DMODEL, pw_bf, DMODEL, pb, hbuf, hbuf, DMODEL, DMODEL, 0, 0);

    ln_kernel<<<Mrows, 256, 0, stream>>>(hbuf, ln2_w + l * DMODEL, ln2_b + l * DMODEL,
                                         ybuf, TPB, TPB, TPB, 1);
    wmma_gemm_bf16<<<dim3(Mrows / 16, MLPDIM / 64), 32, 0, stream>>>(
        ybuf, DMODEL, f1w_bf, DMODEL, f1b, nullptr, mlpb, MLPDIM, DMODEL, 1, 1);
    wmma_gemm_bf16<<<dim3(Mrows / 16, DMODEL / 64), 32, 0, stream>>>(
        mlpb, MLPDIM, f2w_bf, MLPDIM, f2b, hbuf, hbuf, DMODEL, MLPDIM, 0, 0);
  }

  // Final LN writes exactly the (B, 76, 768) fp32 output tensor.
  ln_kernel<<<BATCH * NTOK, 256, 0, stream>>>(hbuf, norm_w, norm_b, (float*)d_out,
                                              NTOK, TPB, NTOK, 0);
}
